// LTNFeedForwardLayer_29678224015512
// MI455X (gfx1250) — matrix operand order repaired and checked by reading, not compiled, not run on hardware
//
#include <hip/hip_runtime.h>

typedef __attribute__((ext_vector_type(16))) _Float16 v16h;
typedef __attribute__((ext_vector_type(8)))  float    v8f;
typedef __attribute__((ext_vector_type(4)))  _Float16 v4h;
typedef __attribute__((ext_vector_type(4)))  unsigned int u32x4;
typedef __attribute__((ext_vector_type(8)))  int i32x8;
typedef __attribute__((ext_vector_type(4)))  int i32x4;

#define B_DIM  4096
#define I_DIM  1024
#define O_DIM  1024
#define N_FAN  4
#define ON_DIM (O_DIM * N_FAN)   // 4096 GEMM columns, c = o*4 + n

#if __has_builtin(__builtin_amdgcn_tensor_load_to_lds) && \
    __has_builtin(__builtin_amdgcn_s_wait_tensorcnt)
#define HAVE_TDM 1
#else
#define HAVE_TDM 0
#endif

// ---------------------------------------------------------------------------
// Kernel 0: convert x (B,I) f32 -> f16 (4 elements per thread, 16B loads)
// ---------------------------------------------------------------------------
__global__ void __launch_bounds__(256)
cvt_f16_kernel(const float* __restrict__ x, _Float16* __restrict__ x16) {
  int idx = (blockIdx.x * 256 + threadIdx.x) * 4;
  float4 f = *(const float4*)(x + idx);
  v4h h;
  h[0] = (_Float16)f.x; h[1] = (_Float16)f.y;
  h[2] = (_Float16)f.z; h[3] = (_Float16)f.w;
  *(v4h*)(x16 + idx) = h;
}

// ---------------------------------------------------------------------------
// Kernel 1: row softmax of mapping_logits (O*N rows, I=1024 each) -> f16
// One wave32 per row; shfl_xor reductions. Output (O*N, I) row-major is
// already the K-contiguous column-major B operand for the GEMM.
// ---------------------------------------------------------------------------
__global__ void __launch_bounds__(256)
softmax_f16_kernel(const float* __restrict__ logits, _Float16* __restrict__ w16) {
  const int row  = blockIdx.x * 8 + (threadIdx.x >> 5);
  const int lane = threadIdx.x & 31;
  const float* rp = logits + (size_t)row * I_DIM;

  float v[32];
  float m = -1e30f;
#pragma unroll
  for (int i = 0; i < 32; ++i) {
    v[i] = rp[lane + i * 32];
    m = fmaxf(m, v[i]);
  }
#pragma unroll
  for (int off = 16; off > 0; off >>= 1)
    m = fmaxf(m, __shfl_xor(m, off, 32));

  float s = 0.f;
#pragma unroll
  for (int i = 0; i < 32; ++i) {
    v[i] = __expf(v[i] - m);
    s += v[i];
  }
#pragma unroll
  for (int off = 16; off > 0; off >>= 1)
    s += __shfl_xor(s, off, 32);

  const float inv = 1.0f / s;
  _Float16* wp = w16 + (size_t)row * I_DIM;
#pragma unroll
  for (int i = 0; i < 32; ++i)
    wp[lane + i * 32] = (_Float16)(v[i] * inv);
}

// ---------------------------------------------------------------------------
// TDM: issue a 2D tile DMA (tile_dim0 = 32 halves contiguous K, `rows` lines,
// line stride 1024 halves) from global f16 into LDS with HW padding that
// produces a 48-half (96B) LDS row stride: 16 DWORDs data + 8 DWORDs pad.
// Descriptor layout per CDNA5 ISA ch.8 (D# groups 0/1; 2D -> groups 2,3 = 0).
// ---------------------------------------------------------------------------
#if HAVE_TDM
__device__ __forceinline__ void tdm_load_tile_f16(unsigned long long gaddr,
                                                  unsigned lds_addr, int rows) {
  u32x4 g0;
  g0[0] = 1u;                                                  // count=1 (valid user D#)
  g0[1] = lds_addr;                                            // LDS byte address
  g0[2] = (unsigned)gaddr;                                     // global_addr[31:0]
  g0[3] = (unsigned)((gaddr >> 32) & 0x01FFFFFFu) | (2u << 30);// addr[56:32] | type=2
  i32x8 g1;
  g1[0] = (int)((1u << 16) |        // data_size = 2 bytes
                (1u << 20) |        // pad_enable
                (3u << 22) |        // pad_interval: 16 DWORDs (=64B = 32 halves)
                (7u << 25));        // pad_amount:    8 DWORDs (=32B = 16 halves)
  g1[1] = (int)(1024u << 16);       // tensor_dim0 = 1024 (bits 79:48, low half)
  g1[2] = 0;                        // tensor_dim0 hi | tensor_dim1 lo
  g1[3] = (int)(16u | (32u << 16)); // tensor_dim1 = 1M (hi bits) | tile_dim0 = 32
  g1[4] = rows;                     // tile_dim1 = rows, tile_dim2 = 0
  g1[5] = 1024;                     // tensor_dim0_stride (elements), low 32
  g1[6] = 0;
  g1[7] = 0;
  i32x4 gz = {0, 0, 0, 0};
#if defined(__clang_major__) && __clang_major__ >= 23
  i32x8 gz8 = {0, 0, 0, 0, 0, 0, 0, 0};
  __builtin_amdgcn_tensor_load_to_lds(g0, g1, gz, gz, gz8, 0);
#else
  __builtin_amdgcn_tensor_load_to_lds(g0, g1, gz, gz, 0);
#endif
}
#endif

// ---------------------------------------------------------------------------
// Kernel 2: fused GEMM (B,I)x(I,O*N) via v_wmma_f32_16x16x32_f16 + sigmoid +
// 16-corner multilinear LUT -> out (B,O) f32.
// Block tile 128(M) x 128(C), K stepped by 32; 8 wave32s in 2(M) x 4(C),
// wave tile 64x32 -> 8 WMMAs per 12 b128 fragment loads per K-step.
// K loop is force-unrolled: buffer parity, LDS offsets and TDM descriptor
// offsets become constants, and accumulator chains stay in place (no phi
// copies). Double-buffered LDS fed by the Tensor Data Mover; TENSORcnt
// completes in order so s_wait_tensorcnt(2) releases the current tile.
// ---------------------------------------------------------------------------
__global__ void __launch_bounds__(256)
gemm_lut_kernel(const _Float16* __restrict__ X16,
                const _Float16* __restrict__ W16,
                const float* __restrict__ lut,
                float* __restrict__ out) {
  constexpr int MT = 128;
  constexpr int NT = 128;
  constexpr int KT = 32;
  constexpr int NKT = I_DIM / KT;       // 32 K-steps
  constexpr int AS = 48;                // padded LDS stride in halves (96B)
  constexpr int FS = 132;               // padded epilogue stride in floats (528B)
  constexpr int A_BYTES = MT * AS * 2;  // 12288
  constexpr int B_BYTES = NT * AS * 2;  // 12288
  constexpr int BUF_BYTES = A_BYTES + B_BYTES;       // 24576 per buffer
  constexpr int SMEM_BYTES = MT * FS * 4;            // 67584 (> 2*BUF_BYTES)

  __shared__ __align__(16) unsigned char smem_raw[SMEM_BYTES];
  __shared__ float lutS[32 * 16];

  const int tid  = threadIdx.x;
  const int lane = tid & 31;
  const int wid  = tid >> 5;
  const int wm   = wid & 1;       // wave M position (x64 rows)
  const int wn   = wid >> 1;      // wave C position (x32 cols)
  const int l16  = lane & 15;
  const int lhi  = lane >> 4;     // 0: lanes 0-15, 1: lanes 16-31

  const int m_base = blockIdx.y * MT;
  const int c_base = blockIdx.x * NT;       // = o_base*4

  // preload this block's 32 LUT rows (32 o x 16 corners = 512 f32)
  lutS[tid]       = lut[blockIdx.x * 512 + tid];
  lutS[tid + 256] = lut[blockIdx.x * 512 + tid + 256];

  v8f acc[4][2] = {};

#if HAVE_TDM
  // LDS byte offset of smem_raw: generic pointer low 32 bits map to LDS addr.
  const unsigned lds0 = (unsigned)(unsigned long long)(void*)smem_raw;
  const unsigned long long gA0 = (unsigned long long)(X16 + (size_t)m_base * I_DIM);
  const unsigned long long gB0 = (unsigned long long)(W16 + (size_t)c_base * I_DIM);
  if (wid == 0) {
    tdm_load_tile_f16(gA0, lds0, MT);
    tdm_load_tile_f16(gB0, lds0 + A_BYTES, NT);
  }
#endif

#pragma unroll
  for (int kt = 0; kt < NKT; ++kt) {
    const int p = kt & 1;
#if HAVE_TDM
    if (wid == 0) {
      if (kt + 1 < NKT) {
        const int kn = (kt + 1) * KT;
        const unsigned lbuf = lds0 + (unsigned)((1 ^ p) * BUF_BYTES);
        tdm_load_tile_f16(gA0 + (size_t)kn * 2, lbuf, MT);
        tdm_load_tile_f16(gB0 + (size_t)kn * 2, lbuf + A_BYTES, NT);
        __builtin_amdgcn_s_wait_tensorcnt(2);  // current tile's 2 DMAs done
      } else {
        __builtin_amdgcn_s_wait_tensorcnt(0);
      }
    }
    __syncthreads();   // current buffer visible to all waves
#else
    // Fallback: register staging, explicit temps so loads stay in flight.
    {
      const int k = kt * KT;
      _Float16* As = (_Float16*)(smem_raw + p * BUF_BYTES);
      _Float16* Bs = (_Float16*)(smem_raw + p * BUF_BYTES + A_BYTES);
      const int id0 = tid, id1 = tid + 256;
      const int r0 = id0 >> 2, q0 = (id0 & 3) * 8;
      const int r1 = id1 >> 2, q1 = (id1 & 3) * 8;
      uint4 ta0 = *(const uint4*)&X16[(size_t)(m_base + r0) * I_DIM + k + q0];
      uint4 ta1 = *(const uint4*)&X16[(size_t)(m_base + r1) * I_DIM + k + q1];
      uint4 tb0 = *(const uint4*)&W16[(size_t)(c_base + r0) * I_DIM + k + q0];
      uint4 tb1 = *(const uint4*)&W16[(size_t)(c_base + r1) * I_DIM + k + q1];
      *(uint4*)&As[r0 * AS + q0] = ta0;
      *(uint4*)&As[r1 * AS + q1] = ta1;
      *(uint4*)&Bs[r0 * AS + q0] = tb0;
      *(uint4*)&Bs[r1 * AS + q1] = tb1;
    }
    __syncthreads();
#endif

    _Float16* As = (_Float16*)(smem_raw + p * BUF_BYTES);
    _Float16* Bs = (_Float16*)(smem_raw + p * BUF_BYTES + A_BYTES);

    // ---- WMMA fragments per the ISA 16-bit A/B VGPR layouts ----
    v16h afrag[4], bfrag[2];
#pragma unroll
    for (int mi = 0; mi < 4; ++mi) {
      const int row = wm * 64 + mi * 16 + l16;
      ((uint4*)&afrag[mi])[0] = *(const uint4*)&As[row * AS + (lhi ? 8 : 0)];
      ((uint4*)&afrag[mi])[1] = *(const uint4*)&As[row * AS + (lhi ? 24 : 16)];
    }
#pragma unroll
    for (int ni = 0; ni < 2; ++ni) {
      const int col = wn * 32 + ni * 16 + l16;
      const int ks  = lhi ? 8 : 0;
      ((uint4*)&bfrag[ni])[0] = *(const uint4*)&Bs[col * AS + ks];
      ((uint4*)&bfrag[ni])[1] = *(const uint4*)&Bs[col * AS + ks + 16];
    }

#pragma unroll
    for (int mi = 0; mi < 4; ++mi)
#pragma unroll
      for (int ni = 0; ni < 2; ++ni)
        acc[mi][ni] = __builtin_amdgcn_wmma_f32_16x16x32_f16(
            false, afrag[mi], false, bfrag[ni],
            (short)0, acc[mi][ni], false, false);

    __syncthreads();   // all waves done reading buffer p; safe to DMA into it
  }

  // ---- epilogue: selected tile -> LDS (f32, padded stride) ----
  float* Sf = (float*)smem_raw;   // [128][132] floats = 67584 B
#pragma unroll
  for (int mi = 0; mi < 4; ++mi)
#pragma unroll
    for (int ni = 0; ni < 2; ++ni) {
      const int col   = wn * 32 + ni * 16 + l16;
      const int rbase = wm * 64 + mi * 16 + lhi * 8;
#pragma unroll
      for (int v = 0; v < 8; ++v)
        Sf[(rbase + v) * FS + col] = acc[mi][ni][v];
    }
  __syncthreads();

  // ---- sigmoid + 16-corner multilinear interpolation, write (B,O) ----
#pragma unroll
  for (int r = 0; r < 16; ++r) {
    const int idx = tid + r * 256;        // 0..4095 = 128 b x 32 o
    const int bl  = idx >> 5;
    const int ol  = idx & 31;
    float4 sv = *(const float4*)&Sf[bl * FS + ol * 4];
    const float s0 = 1.f / (1.f + __expf(-sv.x));
    const float s1 = 1.f / (1.f + __expf(-sv.y));
    const float s2 = 1.f / (1.f + __expf(-sv.z));
    const float s3 = 1.f / (1.f + __expf(-sv.w));
    const float* lp = &lutS[ol * 16];
    float res = 0.f;
#pragma unroll
    for (int c = 0; c < 16; ++c) {
      float w = ((c & 1) ? s0 : 1.f - s0);
      w *= ((c & 2) ? s1 : 1.f - s1);
      w *= ((c & 4) ? s2 : 1.f - s2);
      w *= ((c & 8) ? s3 : 1.f - s3);
      res = fmaf(w, lp[c], res);
    }
    out[(size_t)(m_base + bl) * O_DIM + blockIdx.x * 32 + ol] = res;
  }
}

// ---------------------------------------------------------------------------
extern "C" void kernel_launch(void* const* d_in, const int* in_sizes, int n_in,
                              void* d_out, int out_size, void* d_ws, size_t ws_size,
                              hipStream_t stream) {
  (void)in_sizes; (void)n_in; (void)out_size; (void)ws_size;
  const float* x   = (const float*)d_in[0];   // (B, I)
  const float* ml  = (const float*)d_in[1];   // (O, N, I)
  const float* lut = (const float*)d_in[2];   // (O, 16)
  float* out = (float*)d_out;                 // (B, O)

  _Float16* X16 = (_Float16*)d_ws;                      // 8 MB
  _Float16* W16 = X16 + (size_t)B_DIM * I_DIM;          // 8 MB

  cvt_f16_kernel<<<(B_DIM * I_DIM) / (256 * 4), 256, 0, stream>>>(x, X16);
  softmax_f16_kernel<<<ON_DIM / 8, 256, 0, stream>>>(ml, W16);

  dim3 grid(ON_DIM / 128, B_DIM / 128);  // 32 x 32 = 1024 blocks
  gemm_lut_kernel<<<grid, 256, 0, stream>>>(X16, W16, lut, out);
}
